// CNF_31490700214905
// MI455X (gfx1250) — matrix operand order repaired and checked by reading, not compiled, not run on hardware
//
#include <hip/hip_runtime.h>

typedef _Float16 v8h  __attribute__((ext_vector_type(8)));
typedef _Float16 v16h __attribute__((ext_vector_type(16)));
typedef float    v8f  __attribute__((ext_vector_type(8)));

#define NBLK    2
#define NSTEP   10
#define K0P     160            // layer-0 K padded (1 + 128 -> 160, mult of 32)
#define GSTR    520            // padded row stride (halves) for G buffers (bank-conflict-free)

// f16 weight workspace layout per block, in halves
#define OFF_W1T 0              // [512][512]  W1T[v][u] = W1[u][v]
#define OFF_W2T 262144         // [512][512]
#define OFF_W3T 524288         // [128][512]  W3T[i][v] = W3[v][i]
#define OFF_W0T 589824         // [512][160]  W0T[v][a] = W0[a][v] (a<129), else 0
#define OFF_W0Y 671744         // [128][512]  W0Y[i][u] = W0[1+i][u]
#define BH      737280         // halves per block

// LDS layout (bytes)
#define SM_GTA  0              // 128 x GSTR f16 = 133120
#define SM_GTB  133120
#define SM_H0   266240         // 512 f32
#define SM_H1   268288
#define SM_H2   270336
#define SM_KY   272384         // 6 x 128 f32
#define SM_YV   275456         // 128 f32
#define SM_RED  275968         // 16 f32
#define SM_KL   276032         // 6 f32
#define SM_LP   276056         // 1 f32 (+pad)
#define SM_HINH 276064         // 160 f16 (layer-0 input, f16)
#define SM_H0H  276384         // 512 f16
#define SM_H1H  277408         // 512 f16
#define SM_H2H  278432         // 512 f16
#define SM_ZERO 279456         // 512 f16 of zeros (B-frag source for inactive lanes)
#define SMEM_BYTES 280512

// dt * Butcher coefficients (double products folded to f32, matching jnp.float32(DT*...))
__constant__ float c_DA[6][5] = {
  {0.f,0.f,0.f,0.f,0.f},
  {(float)(-0.1*0.161),0.f,0.f,0.f,0.f},
  {(float)(-0.1*-0.008480655492356989),(float)(-0.1*0.335480655492357),0.f,0.f,0.f},
  {(float)(-0.1*2.8971530571054935),(float)(-0.1*-6.359448489975075),(float)(-0.1*4.3622954328695815),0.f,0.f},
  {(float)(-0.1*5.325864828439257),(float)(-0.1*-11.748883564062828),(float)(-0.1*7.4955393428898365),(float)(-0.1*-0.09249506636175525),0.f},
  {(float)(-0.1*5.86145544294642),(float)(-0.1*-12.92096931784711),(float)(-0.1*8.159367898576159),(float)(-0.1*-0.071584973281401),(float)(-0.1*-0.028269050394068383)}
};
__constant__ float c_DB[6] = {
  (float)(-0.1*0.09646076681806523),(float)(-0.1*0.01),(float)(-0.1*0.4798896504144996),
  (float)(-0.1*1.379008574103742),(float)(-0.1*-3.290069515436081),(float)(-0.1*2.324710524099774)
};
__constant__ float c_DC[6] = {
  0.f,(float)(0.161*-0.1),(float)(0.327*-0.1),(float)(0.9*-0.1),
  (float)(0.9800255409045097*-0.1),(float)(1.0*-0.1)
};

__device__ __forceinline__ v16h cat8(v8h lo, v8h hi) {
  return __builtin_shufflevector(lo, hi, 0,1,2,3,4,5,6,7,8,9,10,11,12,13,14,15);
}
__device__ __forceinline__ v8f wmma16(v16h a, v16h b, v8f c) {
  return __builtin_amdgcn_wmma_f32_16x16x32_f16(false, a, false, b, (short)0, c, false, false);
}
// A fragment 16x32 f16: lanes<16: K {k0..k0+7, k0+16..k0+23}; lanes>=16: +8
__device__ __forceinline__ v16h load_a16(const _Float16* __restrict__ Arow, int k0, int hi8) {
  v8h lo = *(const v8h*)(Arow + k0 + hi8);
  v8h hi = *(const v8h*)(Arow + k0 + 16 + hi8);
  return cat8(lo, hi);
}
__device__ __forceinline__ float wave_reduce(float v) {
  v += __shfl_down(v, 16, 32);
  v += __shfl_down(v, 8, 32);
  v += __shfl_down(v, 4, 32);
  v += __shfl_down(v, 2, 32);
  v += __shfl_down(v, 1, 32);
  return v;
}

// forward layer via WMMA: out[m] = act( sum_k WT[m][k]*hin[k] + bias[m] ).
// B holds the activation vector in column n==0; inactive lanes stream from a zero LDS region
// (single pointer cndmask, no EXEC divergence in the hot loop).
template<bool RELU, bool MIRROR>
__device__ __forceinline__ void fwd_layer(const _Float16* __restrict__ WT, int Kdim,
                                          const _Float16* hin16, const _Float16* zbuf,
                                          const float* __restrict__ bias,
                                          float* hout, _Float16* hout16,
                                          int Mtiles, int wave, int lane) {
  const int nlo = lane & 15;
  const int hi8 = (lane & 16) ? 8 : 0;
  const int k16 = (lane & 16) ? 16 : 0;
  const _Float16* bsrc = (nlo == 0) ? hin16 : zbuf;   // one v_cndmask on the LDS address
  for (int mt = wave; mt < Mtiles; mt += 16) {
    v8f acc = {0.f,0.f,0.f,0.f,0.f,0.f,0.f,0.f};
    const _Float16* Arow = WT + (size_t)(mt*16 + nlo) * (size_t)Kdim;
    for (int k0 = 0; k0 < Kdim; k0 += 32) {
      v16h a = load_a16(Arow, k0, hi8);
      const _Float16* bp = bsrc + k0 + hi8;
      v8h blo = *(const v8h*)(bp);
      v8h bhi = *(const v8h*)(bp + 16);
      acc = wmma16(a, cat8(blo, bhi), acc);
    }
    if (nlo == 0) {
      int mbase = mt*16 + hi8;
      v8h o16;
      #pragma unroll
      for (int r = 0; r < 8; ++r) {
        float v = acc[r] + bias[mbase + r];
        if (RELU) v = fmaxf(v, 0.f);
        hout[mbase + r] = v;
        if (MIRROR) o16[r] = (_Float16)v;
      }
      if (MIRROR) *(v8h*)(hout16 + mbase) = o16;
    }
  }
}

// Ct[n][m] = relu'(h[m]) * sum_k WT[m][k] * Bt[n][k]   (C = 512x128, M=512, N=128, K=512)
__device__ __forceinline__ void jac_gemm(const _Float16* __restrict__ WT,
                                         const _Float16* Bt, _Float16* Ct,
                                         const float* hmask, int wave, int lane) {
  const int nlo = lane & 15;
  const int hi8 = (lane & 16) ? 8 : 0;
  const int k16 = (lane & 16) ? 16 : 0;
  for (int pass = 0; pass < 2; ++pass) {
    const int mt = pass * 16 + wave;
    const _Float16* Arow = WT + (size_t)(mt*16 + nlo) * 512;
    v8f acc[8] = {};
    for (int k0 = 0; k0 < 512; k0 += 32) {
      v16h a = load_a16(Arow, k0, hi8);
      #pragma unroll
      for (int nt = 0; nt < 8; ++nt) {
        const _Float16* bp = Bt + (size_t)(nt*16 + nlo)*GSTR + k0 + hi8;
        v8h blo = *(const v8h*)(bp);
        v8h bhi = *(const v8h*)(bp + 16);
        acc[nt] = wmma16(a, cat8(blo, bhi), acc[nt]);
      }
    }
    const int mbase = mt*16 + hi8;
    float mk[8];
    #pragma unroll
    for (int r = 0; r < 8; ++r) mk[r] = hmask[mbase + r];   // hoisted: reused by all 8 nt
    #pragma unroll
    for (int nt = 0; nt < 8; ++nt) {
      v8h o;
      #pragma unroll
      for (int r = 0; r < 8; ++r) {
        float v = (mk[r] > 0.f) ? acc[nt][r] : 0.f;
        o[r] = (_Float16)v;
      }
      *(v8h*)(Ct + (size_t)(nt*16 + nlo)*GSTR + mbase) = o;
    }
  }
}

__global__ void convert_weights(const float* __restrict__ Ws0, const float* __restrict__ Ws1,
                                const float* __restrict__ Ws2, const float* __restrict__ Ws3,
                                _Float16* __restrict__ wh) {
  int idx = blockIdx.x * 256 + threadIdx.x;
  if (idx >= 2*BH) return;
  int b = idx / BH;
  int r = idx % BH;
  float v;
  if (r < OFF_W2T) {            // W1T [512][512]
    int q = r, vv = q >> 9, u = q & 511;
    v = Ws1[b*262144 + u*512 + vv];
  } else if (r < OFF_W3T) {     // W2T
    int q = r - OFF_W2T, vv = q >> 9, u = q & 511;
    v = Ws2[b*262144 + u*512 + vv];
  } else if (r < OFF_W0T) {     // W3T [128][512]
    int q = r - OFF_W3T, i = q >> 9, vv = q & 511;
    v = Ws3[b*65536 + vv*128 + i];
  } else if (r < OFF_W0Y) {     // W0T [512][160], K padded with zeros
    int q = r - OFF_W0T, vv = q / 160, a = q % 160;
    v = (a < 129) ? Ws0[b*66048 + a*512 + vv] : 0.f;
  } else {                      // W0Y [128][512]
    int q = r - OFF_W0Y, i = q >> 9, u = q & 511;
    v = Ws0[b*66048 + (1 + i)*512 + u];
  }
  wh[idx] = (_Float16)v;
}

__launch_bounds__(512)
__global__ void cnf_kernel(const float* __restrict__ yin,
                           const float* __restrict__ bs0, const float* __restrict__ bs1,
                           const float* __restrict__ bs2, const float* __restrict__ bs3,
                           const _Float16* __restrict__ wh, float* __restrict__ out) {
  extern __shared__ char smem[];
  _Float16* GtA  = (_Float16*)(smem + SM_GTA);
  _Float16* GtB  = (_Float16*)(smem + SM_GTB);
  float* h0   = (float*)(smem + SM_H0);
  float* h1   = (float*)(smem + SM_H1);
  float* h2   = (float*)(smem + SM_H2);
  float* ky   = (float*)(smem + SM_KY);
  float* yv   = (float*)(smem + SM_YV);
  float* red  = (float*)(smem + SM_RED);
  float* kl   = (float*)(smem + SM_KL);
  float* lps  = (float*)(smem + SM_LP);
  _Float16* hinh = (_Float16*)(smem + SM_HINH);
  _Float16* h0h  = (_Float16*)(smem + SM_H0H);
  _Float16* h1h  = (_Float16*)(smem + SM_H1H);
  _Float16* h2h  = (_Float16*)(smem + SM_H2H);
  _Float16* zbuf = (_Float16*)(smem + SM_ZERO);

  const int tid  = threadIdx.x;
  const int wave = tid >> 5;
  const int lane = tid & 31;
  const int s    = blockIdx.x;

  if (tid < 128) yv[tid] = yin[s*128 + tid];
  if (tid >= 129 && tid < 160) hinh[tid] = (_Float16)0.f;  // zero-pad layer-0 input tail
  if (tid < 256) ((unsigned int*)zbuf)[tid] = 0u;          // 1KB zero region
  if (tid == 0) lps[0] = 0.f;
  __syncthreads();

  for (int b = 0; b < NBLK; ++b) {
    const _Float16* W1T = wh + (size_t)b*BH + OFF_W1T;
    const _Float16* W2T = wh + (size_t)b*BH + OFF_W2T;
    const _Float16* W3T = wh + (size_t)b*BH + OFF_W3T;
    const _Float16* W0T = wh + (size_t)b*BH + OFF_W0T;
    const _Float16* W0Y = wh + (size_t)b*BH + OFF_W0Y;
    const float* b0 = bs0 + b*512;
    const float* b1 = bs1 + b*512;
    const float* b2 = bs2 + b*512;
    const float* b3 = bs3 + b*128;

    for (int step = 0; step < NSTEP; ++step) {
      const float tb = 1.0f + (-0.1f) * (float)step;
      for (int stage = 0; stage < 6; ++stage) {
        // --- stage input yi = y + sum_j dt*A[i][j]*ky[j] ; hinh = f16([t, yi, 0...]) ---
        if (tid < 128) {
          float yi = yv[tid];
          for (int j = 0; j < stage; ++j) yi += c_DA[stage][j] * ky[j*128 + tid];
          hinh[1 + tid] = (_Float16)yi;
        }
        if (tid == 0) hinh[0] = (_Float16)(tb + c_DC[stage]);
        __syncthreads();

        // --- forward MLP (WMMA, f32 accumulate) ---
        fwd_layer<true ,true >(W0T, K0P, hinh, zbuf, b0, h0, h0h, 32, wave, lane); __syncthreads();
        fwd_layer<true ,true >(W1T, 512, h0h,  zbuf, b1, h1, h1h, 32, wave, lane); __syncthreads();
        fwd_layer<true ,true >(W2T, 512, h1h,  zbuf, b2, h2, h2h, 32, wave, lane); __syncthreads();
        fwd_layer<false,false>(W3T, 512, h2h,  zbuf, b3, ky + stage*128, (_Float16*)0, 8, wave, lane);

        // --- build G0t[i][u] = relu'(h0[u]) * W0[1+i][u]  (overlaps with layer-3 tail) ---
        {
          const int i = tid >> 2, u0 = (tid & 3) * 128;
          const _Float16* wr = W0Y + (size_t)i*512 + u0;
          const _Float16* hm = h0h + u0;
          _Float16* gr = GtA + (size_t)i*GSTR + u0;
          #pragma unroll 4
          for (int j = 0; j < 128; j += 8) {
            v8h wv = *(const v8h*)(wr + j);
            v8h hv = *(const v8h*)(hm + j);
            v8h g;
            #pragma unroll
            for (int e = 0; e < 8; ++e)
              g[e] = (hv[e] > (_Float16)0.f) ? wv[e] : (_Float16)0.f;
            *(v8h*)(gr + j) = g;
          }
        }
        __syncthreads();

        // --- Jacobian chain: G1 = D1 W1^T G0 ; G2 = D2 W2^T G1 ---
        jac_gemm(W1T, GtA, GtB, h1, wave, lane); __syncthreads();
        jac_gemm(W2T, GtB, GtA, h2, wave, lane); __syncthreads();

        // --- trace = sum_{i,v} W3T[i][v] * G2t[i][v] ---
        {
          const int i = tid >> 2, v0 = (tid & 3) * 128;
          const _Float16* w3r = W3T + (size_t)i*512 + v0;
          const _Float16* g2r = GtA + (size_t)i*GSTR + v0;
          float acc = 0.f;
          #pragma unroll 4
          for (int j = 0; j < 128; j += 8) {
            v8h a = *(const v8h*)(w3r + j);
            v8h g = *(const v8h*)(g2r + j);
            #pragma unroll
            for (int e = 0; e < 8; ++e) acc += (float)a[e] * (float)g[e];
          }
          acc = wave_reduce(acc);
          if (lane == 0) red[wave] = acc;
        }
        __syncthreads();
        if (wave == 0) {
          float v = (lane < 16) ? red[lane] : 0.f;
          v = wave_reduce(v);
          if (lane == 0) kl[stage] = v;
        }
        __syncthreads();
      }
      // --- combine: y += sum dt*b_i*ky_i ; lp += sum dt*b_i*klp_i ---
      if (tid < 128) {
        float a = yv[tid];
        #pragma unroll
        for (int i2 = 0; i2 < 6; ++i2) a += c_DB[i2] * ky[i2*128 + tid];
        yv[tid] = a;
      }
      if (tid == 0) {
        float a = lps[0];
        #pragma unroll
        for (int i2 = 0; i2 < 6; ++i2) a += c_DB[i2] * kl[i2];
        lps[0] = a;
      }
      __syncthreads();
    }
  }
  // out = lp - 0.5*(log(2*pi) + sum(y^2))
  if (wave == 0) {
    float ss = 0.f;
    for (int j = lane; j < 128; j += 32) ss += yv[j] * yv[j];
    ss = wave_reduce(ss);
    if (lane == 0) out[s] = lps[0] - 0.5f * (1.8378770664093453f + ss);
  }
}

extern "C" void kernel_launch(void* const* d_in, const int* in_sizes, int n_in,
                              void* d_out, int out_size, void* d_ws, size_t ws_size,
                              hipStream_t stream) {
  const float* y   = (const float*)d_in[0];
  const float* Ws0 = (const float*)d_in[1];
  const float* bs0 = (const float*)d_in[2];
  const float* Ws1 = (const float*)d_in[3];
  const float* bs1 = (const float*)d_in[4];
  const float* Ws2 = (const float*)d_in[5];
  const float* bs2 = (const float*)d_in[6];
  const float* Ws3 = (const float*)d_in[7];
  const float* bs3 = (const float*)d_in[8];
  _Float16* wh = (_Float16*)d_ws;
  float* out = (float*)d_out;

  const int total = 2 * BH;
  convert_weights<<<(total + 255) / 256, 256, 0, stream>>>(Ws0, Ws1, Ws2, Ws3, wh);

  (void)hipFuncSetAttribute((const void*)cnf_kernel,
                            hipFuncAttributeMaxDynamicSharedMemorySize, SMEM_BYTES);
  cnf_kernel<<<512, 512, SMEM_BYTES, stream>>>(y, bs0, bs1, bs2, bs3, wh, out);
}
